// DualGraphLearner_25537875542502
// MI455X (gfx1250) — matrix operand order repaired and checked by reading, not compiled, not run on hardware
//
#include <hip/hip_runtime.h>
#include <hip/hip_bf16.h>
#include <stdint.h>

#define NN 10000
#define EE 320000
#define DD 256

typedef __attribute__((ext_vector_type(16))) __bf16 v16bf;
typedef __attribute__((ext_vector_type(8)))  float  v8f;

union Frag16 { uint4 u[2]; v16bf v; };

__device__ __forceinline__ unsigned short f2bf(float f) {
  unsigned u = __float_as_uint(f);
  unsigned r = (u + 0x7FFFu + ((u >> 16) & 1u)) >> 16;   // RNE
  return (unsigned short)r;
}
__device__ __forceinline__ unsigned ordf(float f) {       // order-preserving map
  unsigned u = __float_as_uint(f);
  return (u & 0x80000000u) ? ~u : (u | 0x80000000u);
}
__device__ __forceinline__ float decf(unsigned u) {
  return __uint_as_float((u & 0x80000000u) ? (u & 0x7FFFFFFFu) : ~u);
}
__device__ __forceinline__ float waveReduceSum(float v) {
  #pragma unroll
  for (int off = 16; off > 0; off >>= 1) v += __shfl_xor(v, off, 32);
  return v;
}

// ---------------- prep: convert x to bf16 -----------------------------------
__global__ __launch_bounds__(256) void k_cvt_x(const float* __restrict__ x,
                                               unsigned short* __restrict__ xbf) {
  int i = blockIdx.x * 256 + threadIdx.x;
  if (i < NN * DD) xbf[i] = f2bf(x[i]);
}

// ---------------- prep: swizzle Wcat (256x1024) into WMMA B-fragment order --
// wB[kt][ntile][lane][h]  (kt<8, ntile<64, lane<32, h<16), 32B contiguous/lane
__global__ __launch_bounds__(256) void k_prep_w(
    const float* __restrict__ Wq, const float* __restrict__ Wk,
    const float* __restrict__ Wv, const float* __restrict__ Ws,
    const float* __restrict__ bq, const float* __restrict__ bk,
    const float* __restrict__ bv, const float* __restrict__ bs,
    unsigned short* __restrict__ wB, float* __restrict__ bcat) {
  int idx = blockIdx.x * 256 + threadIdx.x;            // 262144 total
  if (idx < 262144) {
    int h = idx & 15, lane = (idx >> 4) & 31, ntile = (idx >> 9) & 63, kt = idx >> 15;
    int k = kt * 32 + ((lane >> 4) * 16) + h;          // contiguous k order (same as A): lanes 0-15 K=0..15, 16-31 K=16..31
    int n = ntile * 16 + (lane & 15);
    float v;
    if      (n < 256) v = Wq[k * 256 + n];
    else if (n < 512) v = Wk[k * 256 + (n - 256)];
    else if (n < 768) v = Wv[k * 256 + (n - 512)];
    else              v = Ws[k * 256 + (n - 768)];
    wB[idx] = f2bf(v);
  }
  if (idx < 1024) {
    float b;
    if      (idx < 256) b = bq[idx];
    else if (idx < 512) b = bk[idx - 256];
    else if (idx < 768) b = bv[idx - 512];
    else                b = bs[idx - 768];
    bcat[idx] = b;
  }
}

// ---------------- init accumulators -----------------------------------------
__global__ __launch_bounds__(256) void k_init(float* __restrict__ aggr,
                                              float* __restrict__ denom,
                                              unsigned* __restrict__ mbuf,
                                              float* __restrict__ colsum,
                                              float* __restrict__ colsq,
                                              unsigned* __restrict__ minmax) {
  int i = blockIdx.x * 256 + threadIdx.x;
  if (i < NN * DD) aggr[i] = 0.0f;
  if (i < NN) { denom[i] = 0.0f; mbuf[i] = 0x007FFFFFu; }  // ordf(-inf)
  if (i < DD) { colsum[i] = 0.0f; colsq[i] = 0.0f; }
  if (i == 0) { minmax[0] = 0xFFFFFFFFu; minmax[1] = 0u; } // minenc, maxenc
}

// ---------------- fused projection GEMM: [Q|K|V|S] = x @ Wcat + bcat --------
// grid (625, 8), 256 threads = 8 waves; wave w -> ntile = blockIdx.y*8 + w
__global__ __launch_bounds__(256) void k_gemm_qkvs(
    const unsigned short* __restrict__ xbf, const unsigned short* __restrict__ wB,
    const float* __restrict__ bcat, float* __restrict__ qkvs) {
  __shared__ __align__(16) unsigned short lx[16 * 256];   // 8KB x-tile, bf16
  const int lane = threadIdx.x & 31;
  const int wave = threadIdx.x >> 5;
  const int m0   = blockIdx.x * 16;

  // Async-copy the contiguous 8KB tile (rows m0..m0+15 of bf16 x) into LDS.
  {
    unsigned lds = (unsigned)(uintptr_t)(&lx[0]) + threadIdx.x * 16u;
    unsigned long long ga =
        (unsigned long long)(uintptr_t)(xbf + (size_t)m0 * 256) +
        (unsigned long long)threadIdx.x * 16ull;
    asm volatile("global_load_async_to_lds_b128 %0, %1, off"
                 :: "v"(lds), "v"(ga) : "memory");
    asm volatile("global_load_async_to_lds_b128 %0, %1, off offset:4096"
                 :: "v"(lds), "v"(ga) : "memory");
    asm volatile("s_wait_asynccnt 0" ::: "memory");
  }
  __syncthreads();

  const int mrow  = lane & 15;
  const int kb    = (lane >> 4) * 16;       // contiguous k order: lanes 16-31 hold K+16..K+31
  const int ntile = blockIdx.y * 8 + wave;  // 0..63

  v8f acc = {0.f, 0.f, 0.f, 0.f, 0.f, 0.f, 0.f, 0.f};
  #pragma unroll
  for (int kt = 0; kt < 8; ++kt) {
    const int k0 = kt * 32;
    Frag16 A, B;
    A.u[0] = *(const uint4*)&lx[mrow * 256 + k0 + kb];        // halves 0..7 : K=k0+kb..+7
    A.u[1] = *(const uint4*)&lx[mrow * 256 + k0 + kb + 8];    // halves 8..15: K=k0+kb+8..+15
    const unsigned short* bp = wB + ((((size_t)kt * 64 + ntile) * 32 + lane) * 16);
    B.u[0] = *(const uint4*)(bp);
    B.u[1] = *(const uint4*)(bp + 8);
    acc = __builtin_amdgcn_wmma_f32_16x16x32_bf16(false, A.v, false, B.v,
                                                  (short)0, acc, false, false);
  }

  const int ncol  = ntile * 16 + (lane & 15);
  const int mbase = m0 + ((lane >> 4) * 8);
  const float bias = bcat[ncol];
  #pragma unroll
  for (int r = 0; r < 8; ++r)
    qkvs[(size_t)(mbase + r) * 1024 + ncol] = acc[r] + bias;
}

// ---------------- edge pass 1: score + segment max --------------------------
__global__ __launch_bounds__(256) void k_edge_score(
    const long long* __restrict__ ei, const float* __restrict__ qkvs,
    float* __restrict__ score, unsigned* __restrict__ mbuf) {
  int w = (blockIdx.x * 256 + threadIdx.x) >> 5;
  int lane = threadIdx.x & 31;
  if (w >= EE) return;
  int s = (int)ei[w], d = (int)ei[EE + w];
  const float* q = qkvs + (size_t)d * 1024;         // Q at cols 0..255
  const float* k = qkvs + (size_t)s * 1024 + 256;   // K at cols 256..511
  int f = lane * 8;
  float4 q0 = *(const float4*)(q + f), q1 = *(const float4*)(q + f + 4);
  float4 k0 = *(const float4*)(k + f), k1 = *(const float4*)(k + f + 4);
  float p = q0.x*k0.x + q0.y*k0.y + q0.z*k0.z + q0.w*k0.w
          + q1.x*k1.x + q1.y*k1.y + q1.z*k1.z + q1.w*k1.w;
  p = waveReduceSum(p);
  if (lane == 0) {
    float sc = p * 0.0625f;                          // 1/sqrt(256)
    score[w] = sc;
    atomicMax(mbuf + d, ordf(sc));
  }
}

// ---------------- edge pass 2: exp + segment sum ----------------------------
__global__ __launch_bounds__(256) void k_edge_soft(
    const long long* __restrict__ ei, float* __restrict__ score,
    const unsigned* __restrict__ mbuf, float* __restrict__ denom) {
  int e = blockIdx.x * 256 + threadIdx.x;
  if (e >= EE) return;
  int d = (int)ei[EE + e];
  float a = __expf(score[e] - decf(mbuf[d]));
  score[e] = a;                                      // in-place numerator
  atomicAdd(denom + d, a);
}

// ---------------- edge pass 3: weighted scatter of V ------------------------
__global__ __launch_bounds__(256) void k_edge_aggr(
    const long long* __restrict__ ei, const float* __restrict__ score,
    const float* __restrict__ denom, const float* __restrict__ qkvs,
    float* __restrict__ aggr) {
  int w = (blockIdx.x * 256 + threadIdx.x) >> 5;
  int lane = threadIdx.x & 31;
  if (w >= EE) return;
  int s = (int)ei[w], d = (int)ei[EE + w];
  float alpha = score[w] / denom[d];
  const float* v = qkvs + (size_t)s * 1024 + 512;    // V at cols 512..767
  float* ag = aggr + (size_t)d * 256;
  int f = lane * 8;
  float4 v0 = *(const float4*)(v + f), v1 = *(const float4*)(v + f + 4);
  atomicAdd(ag + f + 0, alpha * v0.x); atomicAdd(ag + f + 1, alpha * v0.y);
  atomicAdd(ag + f + 2, alpha * v0.z); atomicAdd(ag + f + 3, alpha * v0.w);
  atomicAdd(ag + f + 4, alpha * v1.x); atomicAdd(ag + f + 5, alpha * v1.y);
  atomicAdd(ag + f + 6, alpha * v1.z); atomicAdd(ag + f + 7, alpha * v1.w);
}

// ---------------- h = aggr + skip + x ; column sums -------------------------
__global__ __launch_bounds__(256) void k_node_h(
    const float* __restrict__ x, const float* __restrict__ qkvs,
    float* __restrict__ aggr /* becomes h */, float* __restrict__ colsum) {
  int f = threadIdx.x;
  int n0 = blockIdx.x * 64;
  float loc = 0.f;
  for (int i = 0; i < 64; ++i) {
    int n = n0 + i;
    if (n < NN) {
      size_t o = (size_t)n * 256 + f;
      float h = aggr[o] + qkvs[(size_t)n * 1024 + 768 + f] + x[o];
      aggr[o] = h;
      loc += h;
    }
  }
  atomicAdd(colsum + f, loc);
}

__global__ __launch_bounds__(256) void k_stat_mean(const float* __restrict__ colsum,
                                                   float* __restrict__ meanbuf) {
  int f = threadIdx.x;
  meanbuf[f] = colsum[f] * (1.0f / NN);
}

// ---------------- c = h - mean*scale ; column sumsq -------------------------
__global__ __launch_bounds__(256) void k_node_var(
    float* __restrict__ hbuf, const float* __restrict__ meanbuf,
    const float* __restrict__ gms, float* __restrict__ colsq) {
  int f = threadIdx.x;
  int n0 = blockIdx.x * 64;
  float ms = meanbuf[f] * gms[f];
  float loc = 0.f;
  for (int i = 0; i < 64; ++i) {
    int n = n0 + i;
    if (n < NN) {
      size_t o = (size_t)n * 256 + f;
      float c = hbuf[o] - ms;
      hbuf[o] = c;
      loc += c * c;
    }
  }
  atomicAdd(colsq + f, loc);
}

__global__ __launch_bounds__(256) void k_stat_rstd(const float* __restrict__ colsq,
                                                   float* __restrict__ rstd) {
  int f = threadIdx.x;
  rstd[f] = rsqrtf(colsq[f] * (1.0f / NN) + 1e-5f);
}

// ---------------- normalize + ReLU + y = h@Wout ; global min/max ------------
__global__ __launch_bounds__(256) void k_node_y(
    const float* __restrict__ cbuf, const float* __restrict__ rstd,
    const float* __restrict__ gnw, const float* __restrict__ gnb,
    const float* __restrict__ Wout, const float* __restrict__ bout,
    float* __restrict__ ybuf, unsigned* __restrict__ minmax) {
  int n = (blockIdx.x * 256 + threadIdx.x) >> 5;
  int lane = threadIdx.x & 31;
  if (n >= NN) return;
  const float* c = cbuf + (size_t)n * 256;
  float sum = 0.f;
  #pragma unroll
  for (int j = 0; j < 8; ++j) {
    int f = lane * 8 + j;
    float h = gnw[f] * c[f] * rstd[f] + gnb[f];
    h = fmaxf(h, 0.f);
    sum += h * Wout[f];
  }
  sum = waveReduceSum(sum);
  if (lane == 0) {
    float y = sum + bout[0];
    ybuf[n] = y;
    atomicMin(minmax + 0, ordf(y));
    atomicMax(minmax + 1, ordf(y));
  }
}

__global__ __launch_bounds__(256) void k_final(const float* __restrict__ ybuf,
                                               const unsigned* __restrict__ minmax,
                                               float* __restrict__ out) {
  int n = blockIdx.x * 256 + threadIdx.x;
  if (n >= NN) return;
  float ymin = decf(minmax[0]), ymax = decf(minmax[1]);
  out[n] = (ybuf[n] - ymin) / (ymax - ymin + 1e-8f);
}

// ---------------------------------------------------------------------------
extern "C" void kernel_launch(void* const* d_in, const int* in_sizes, int n_in,
                              void* d_out, int out_size, void* d_ws, size_t ws_size,
                              hipStream_t stream) {
  const float*     x    = (const float*)d_in[0];
  const long long* ei   = (const long long*)d_in[1];   // int64 (2,E)
  const float* Wq = (const float*)d_in[2];  const float* bq = (const float*)d_in[3];
  const float* Wk = (const float*)d_in[4];  const float* bk = (const float*)d_in[5];
  const float* Wv = (const float*)d_in[6];  const float* bv = (const float*)d_in[7];
  const float* Ws = (const float*)d_in[8];  const float* bs = (const float*)d_in[9];
  const float* gnw = (const float*)d_in[10];
  const float* gnb = (const float*)d_in[11];
  const float* gms = (const float*)d_in[12];
  const float* Wout = (const float*)d_in[13];
  const float* bout = (const float*)d_in[14];
  float* out = (float*)d_out;

  // workspace carve-out (~60 MB total)
  char* base = (char*)d_ws;
  size_t off = 0;
  auto carve = [&](size_t bytes) -> void* {
    off = (off + 255) & ~(size_t)255;
    void* p = base + off;
    off += bytes;
    return p;
  };
  unsigned short* xbf   = (unsigned short*)carve((size_t)NN * DD * 2);
  unsigned short* wB    = (unsigned short*)carve((size_t)262144 * 2);
  float*          bcat  = (float*)carve(1024 * 4);
  float*          qkvs  = (float*)carve((size_t)NN * 1024 * 4);
  float*          score = (float*)carve((size_t)EE * 4);
  unsigned*       mbuf  = (unsigned*)carve((size_t)NN * 4);
  float*          denom = (float*)carve((size_t)NN * 4);
  float*          aggr  = (float*)carve((size_t)NN * DD * 4);   // reused as h, then c
  float*          colsum = (float*)carve(DD * 4);
  float*          colsq  = (float*)carve(DD * 4);
  float*          meanbuf = (float*)carve(DD * 4);
  float*          rstd    = (float*)carve(DD * 4);
  float*          ybuf    = (float*)carve((size_t)NN * 4);
  unsigned*       minmax  = (unsigned*)carve(2 * 4);

  k_cvt_x<<<(NN * DD + 255) / 256, 256, 0, stream>>>(x, xbf);
  k_prep_w<<<1024, 256, 0, stream>>>(Wq, Wk, Wv, Ws, bq, bk, bv, bs, wB, bcat);
  k_init<<<(NN * DD + 255) / 256, 256, 0, stream>>>(aggr, denom, mbuf, colsum, colsq, minmax);

  k_gemm_qkvs<<<dim3(NN / 16, 8), 256, 0, stream>>>(xbf, wB, bcat, qkvs);

  k_edge_score<<<(EE * 32 + 255) / 256, 256, 0, stream>>>(ei, qkvs, score, mbuf);
  k_edge_soft<<<(EE + 255) / 256, 256, 0, stream>>>(ei, score, mbuf, denom);
  k_edge_aggr<<<(EE * 32 + 255) / 256, 256, 0, stream>>>(ei, score, denom, qkvs, aggr);

  k_node_h<<<(NN + 63) / 64, 256, 0, stream>>>(x, qkvs, aggr, colsum);
  k_stat_mean<<<1, 256, 0, stream>>>(colsum, meanbuf);
  k_node_var<<<(NN + 63) / 64, 256, 0, stream>>>(aggr, meanbuf, gms, colsq);
  k_stat_rstd<<<1, 256, 0, stream>>>(colsq, rstd);

  k_node_y<<<(NN * 32 + 255) / 256, 256, 0, stream>>>(aggr, rstd, gnw, gnb, Wout, bout, ybuf, minmax);
  k_final<<<(NN + 255) / 256, 256, 0, stream>>>(ybuf, minmax, out);
}
